// ConvAttention_45603962749204
// MI455X (gfx1250) — matrix operand order repaired and checked by reading, not compiled, not run on hardware
//
#include <hip/hip_runtime.h>
#include <hip/hip_bf16.h>

// ---------------------------------------------------------------------------
// ConvAttention forward for MI455X (gfx1250, wave32, WMMA).
// Pipeline:
//   1) Q/K/V = q @ W{q,k,v}          (bf16 WMMA 16x16x32, fp32 acc)
//   2) kc/vc = depthwise conv stride4 (elementwise)
//   3) logits = qh @ kc^T  -> written raw to d_out (output tensor)
//      online softmax + att@vc fused (flash-style, logits written ONCE)
//   4) out = attn @ Wlin   (fp32 out)
// ---------------------------------------------------------------------------

typedef __bf16 bf16;
typedef __attribute__((ext_vector_type(16))) __bf16 v16bf;
typedef __attribute__((ext_vector_type(8)))  float  v8f;

#define B_    2
#define S_    4096
#define HID_  1024
#define QKV_  1024
#define H_    16
#define HD_   64
#define SK_   1024   // (S-4)/4 + 1

__device__ __forceinline__ v8f zero_v8f() {
  v8f z;
#pragma unroll
  for (int i = 0; i < 8; ++i) z[i] = 0.0f;
  return z;
}

__device__ __forceinline__ v8f wmma_bf16(v16bf a, v16bf b, v8f c) {
  return __builtin_amdgcn_wmma_f32_16x16x32_bf16(
      /*neg_a=*/false, a, /*neg_b=*/false, b,
      /*c_mod=*/(short)0, c, /*reuse_a=*/false, /*reuse_b=*/false);
}

// A fragment: 16x32 bf16, row-major source in LDS.
// lane (0-15) holds row lane, Ks {0..7,16..23}; lane+16 holds row lane,
// Ks {8..15,24..31}  (ISA 7.12.2, 16-bit A 16x32 layout).
__device__ __forceinline__ v16bf lds_load_a_frag(const bf16* base, int row_base,
                                                 int lda, int koff) {
  const int lane = threadIdx.x & 31;
  const int half = lane >> 4;
  const bf16* rp = base + (size_t)(row_base + (lane & 15)) * lda + koff;
  v16bf a;
#pragma unroll
  for (int v = 0; v < 8; ++v) {
    const int kb = ((v & 4) << 2) + (half << 3) + ((v & 3) << 1);
    a[2 * v]     = rp[kb];
    a[2 * v + 1] = rp[kb + 1];
  }
  return a;
}

// B fragment: 32x16 bf16 (KxN). Source in LDS stored N-major: base[n][k].
// lanes 0-15 hold K {0..7,16..23} of column lane; lanes 16-31 hold K {8..15,24..31}.
__device__ __forceinline__ v16bf lds_load_b_frag(const bf16* base, int n_base,
                                                 int ldn, int koff) {
  const int lane = threadIdx.x & 31;
  const bf16* cp = base + (size_t)(n_base + (lane & 15)) * ldn + koff +
                   ((lane >> 4) << 3);
  v16bf b;
#pragma unroll
  for (int e = 0; e < 16; ++e) b[e] = cp[((e & 8) << 1) + (e & 7)];
  return b;
}

// ---------------------------------------------------------------------------
// Generic 128x128 tiled GEMM: C[M,N] = A[M,K] @ W[K,N]
//   A: fp32 or bf16 (template), W: fp32 (converted to bf16 on LDS stage)
//   out: bf16 (optionally scaled by exp(-log_temp)) or fp32
// block = 256 threads (8 waves); wave w owns a 16x128 strip of C.
// ---------------------------------------------------------------------------
template <bool A_IS_F32, bool OUT_F32>
__global__ __launch_bounds__(256) void gemm_bf16_kernel(
    const void* __restrict__ Aptr, const float* __restrict__ W,
    bf16* __restrict__ outB, float* __restrict__ outF,
    int M, int N, int K, const float* __restrict__ logt) {
  __shared__ bf16 As[128 * 32];   // 8 KB, row-major [m][k]
  __shared__ bf16 Bs[128 * 32];   // 8 KB, transposed [n][k]
  const int t    = threadIdx.x;
  const int wave = t >> 5;
  const int lane = t & 31;
  const int mb   = blockIdx.x * 128;
  const int nb   = blockIdx.y * 128;

  const float scale = logt ? __expf(-logt[0]) : 1.0f;

  v8f acc[8];
#pragma unroll
  for (int j = 0; j < 8; ++j) acc[j] = zero_v8f();

  for (int kt = 0; kt < K; kt += 32) {
    // ---- stage A tile (128x32) : 16 elems/thread, contiguous in K ----
    {
      const int row = t >> 1, k0 = (t & 1) * 16;
      if (A_IS_F32) {
        const float* src = (const float*)Aptr + (size_t)(mb + row) * K + kt + k0;
#pragma unroll
        for (int i = 0; i < 16; ++i) As[row * 32 + k0 + i] = (bf16)src[i];
        if (kt + 32 < K) __builtin_prefetch(src + 32, 0, 1);
      } else {
        const bf16* src = (const bf16*)Aptr + (size_t)(mb + row) * K + kt + k0;
#pragma unroll
        for (int i = 0; i < 16; ++i) As[row * 32 + k0 + i] = src[i];
        if (kt + 32 < K) __builtin_prefetch(src + 32, 0, 1);
      }
    }
    // ---- stage B tile transposed: read W rows contiguously, write [n][k] --
    {
      const int k = t >> 3, n0 = (t & 7) * 16;
      const float* src = W + (size_t)(kt + k) * N + nb + n0;
#pragma unroll
      for (int i = 0; i < 16; ++i) Bs[(n0 + i) * 32 + k] = (bf16)src[i];
      if (kt + 32 < K) __builtin_prefetch(src + (size_t)32 * N, 0, 1);
    }
    __syncthreads();

    const v16bf a = lds_load_a_frag(As, wave * 16, 32, 0);
#pragma unroll
    for (int j = 0; j < 8; ++j) {
      const v16bf b = lds_load_b_frag(Bs, j * 16, 32, 0);
      acc[j] = wmma_bf16(a, b, acc[j]);
    }
    __syncthreads();
  }

  // ---- epilogue: C layout = lane holds col (lane&15), rows v + 8*(lane>>4)
  const int rsel = 8 * (lane >> 4);
  const int col  = lane & 15;
#pragma unroll
  for (int j = 0; j < 8; ++j) {
#pragma unroll
    for (int v = 0; v < 8; ++v) {
      const size_t r = (size_t)mb + wave * 16 + v + rsel;
      const size_t c = (size_t)nb + j * 16 + col;
      const float val = acc[j][v] * scale;
      if (OUT_F32) outF[r * N + c] = val;
      else         outB[r * N + c] = (bf16)val;
    }
  }
}

// ---------------------------------------------------------------------------
// Depthwise conv, kernel (4,1) stride (4,1): kc[b,h,sk,d] = sum_i K[b,4sk+i,h*64+d]*w[h,i]
// ---------------------------------------------------------------------------
__global__ __launch_bounds__(256) void dwconv_kernel(
    const bf16* __restrict__ Kin, const bf16* __restrict__ Vin,
    const float* __restrict__ cw, bf16* __restrict__ kc,
    bf16* __restrict__ vc) {
  const int o = blockIdx.x * 256 + threadIdx.x;  // [0, B*H*SK*HD)
  const int d  = o & 63;
  const int sk = (o >> 6) & (SK_ - 1);
  const int h  = (o >> 16) & (H_ - 1);
  const int b  = o >> 20;
  float ak = 0.f, av = 0.f;
  const size_t base = ((size_t)b * S_ + 4 * sk) * QKV_ + h * 64 + d;
#pragma unroll
  for (int i = 0; i < 4; ++i) {
    const float w = cw[h * 4 + i];
    ak += w * (float)Kin[base + (size_t)i * QKV_];
    av += w * (float)Vin[base + (size_t)i * QKV_];
  }
  kc[o] = (bf16)ak;
  vc[o] = (bf16)av;
}

// ---------------------------------------------------------------------------
// Fused attention: per (b,h), 128-query tile per block (8 waves x 16 rows).
// Loop over 64-key chunks: logits via WMMA -> raw store to d_out, online
// softmax (shfl reductions across the 16-lane half owning each row), P
// re-laid-out through per-wave LDS into A-fragments, P @ vc^T via WMMA.
// ---------------------------------------------------------------------------
__global__ __launch_bounds__(256) void attn_kernel(
    const bf16* __restrict__ Q, const bf16* __restrict__ kc,
    const bf16* __restrict__ vc, float* __restrict__ logits,
    bf16* __restrict__ attn_out) {
  __shared__ bf16 qs[128 * 64];    // 16 KB : q tile, row-major [row][d]
  __shared__ bf16 ks[64 * 64];     //  8 KB : kc chunk, [key][d]  (B for GEMM1)
  __shared__ bf16 vts[64 * 64];    //  8 KB : vc chunk transposed [d][key] (B for GEMM2)
  __shared__ bf16 ps[8][16 * 64];  // 16 KB : per-wave P tile [16][64]

  const int t    = threadIdx.x;
  const int wave = t >> 5;
  const int lane = t & 31;
  const int qb   = blockIdx.x * 128;
  const int bh   = blockIdx.y;          // b*H + h
  const int b    = bh >> 4, h = bh & 15;

  // ---- stage q tile (already temperature-scaled in projection) ----
  {
    const int row = t >> 1, seg = (t & 1) * 32;
    const bf16* src = Q + ((size_t)b * S_ + qb + row) * QKV_ + h * 64 + seg;
#pragma unroll
    for (int i = 0; i < 32; ++i) qs[row * 64 + seg + i] = src[i];
  }
  __syncthreads();
  const v16bf a_lo = lds_load_a_frag(qs, wave * 16, 64, 0);
  const v16bf a_hi = lds_load_a_frag(qs, wave * 16, 64, 32);

  float m[8], ssum[8];
#pragma unroll
  for (int v = 0; v < 8; ++v) { m[v] = -3.0e38f; ssum[v] = 0.f; }
  v8f oacc[4];
#pragma unroll
  for (int j = 0; j < 4; ++j) oacc[j] = zero_v8f();

  const size_t kvbase = (size_t)bh * SK_ * HD_;
  float* lrow = logits + ((size_t)bh * S_ + qb) * SK_;
  const int rsel = 8 * (lane >> 4);
  const int col  = lane & 15;

  for (int kt = 0; kt < SK_; kt += 64) {
    // ---- stage kc chunk (straight copy) and vc chunk (transposed) ----
    {
      const int idx = t * 16;
      const bf16* src = kc + kvbase + (size_t)kt * HD_ + idx;
#pragma unroll
      for (int i = 0; i < 16; ++i) ks[idx + i] = src[i];
      if (kt + 64 < SK_) __builtin_prefetch(src + 64 * HD_, 0, 1);
    }
    {
      const int key = t >> 2, d0 = (t & 3) * 16;
      const bf16* src = vc + kvbase + (size_t)(kt + key) * HD_ + d0;
#pragma unroll
      for (int i = 0; i < 16; ++i) vts[(d0 + i) * 64 + key] = src[i];
      if (kt + 64 < SK_) __builtin_prefetch(src + 64 * HD_, 0, 1);
    }
    __syncthreads();

    // ---- logits tiles: 16x64 per wave, K = HD = 64 (two WMMAs per tile) ----
    v8f cacc[4];
#pragma unroll
    for (int jn = 0; jn < 4; ++jn) {
      v8f c = zero_v8f();
      c = wmma_bf16(a_lo, lds_load_b_frag(ks, jn * 16, 64, 0), c);
      c = wmma_bf16(a_hi, lds_load_b_frag(ks, jn * 16, 64, 32), c);
      cacc[jn] = c;
    }

    // ---- store RAW logits (mandatory output) — written exactly once ----
#pragma unroll
    for (int jn = 0; jn < 4; ++jn)
#pragma unroll
      for (int v = 0; v < 8; ++v)
        lrow[(size_t)(wave * 16 + v + rsel) * SK_ + kt + jn * 16 + col] =
            cacc[jn][v];

    // ---- online softmax: row r = v + rsel lives on 16 lanes of a half ----
#pragma unroll
    for (int v = 0; v < 8; ++v) {
      float cm = fmaxf(fmaxf(cacc[0][v], cacc[1][v]),
                       fmaxf(cacc[2][v], cacc[3][v]));
#pragma unroll
      for (int msk = 8; msk >= 1; msk >>= 1)
        cm = fmaxf(cm, __shfl_xor(cm, msk, 32));
      const float nm    = fmaxf(m[v], cm);
      const float alpha = __expf(m[v] - nm);
      m[v] = nm;
      float rs = 0.f;
#pragma unroll
      for (int jn = 0; jn < 4; ++jn) {
        const float p = __expf(cacc[jn][v] - nm);
        cacc[jn][v] = p;
        rs += p;
      }
#pragma unroll
      for (int msk = 8; msk >= 1; msk >>= 1)
        rs += __shfl_xor(rs, msk, 32);
      ssum[v] = ssum[v] * alpha + rs;
#pragma unroll
      for (int j = 0; j < 4; ++j) oacc[j][v] *= alpha;
    }

    // ---- P: C-layout -> A-fragment layout via per-wave LDS ----
    {
      bf16* pw = &ps[wave][0];
#pragma unroll
      for (int jn = 0; jn < 4; ++jn)
#pragma unroll
        for (int v = 0; v < 8; ++v)
          pw[(v + rsel) * 64 + jn * 16 + col] = (bf16)cacc[jn][v];
    }
    asm volatile("s_wait_dscnt 0" ::: "memory");  // wave-private RAW through LDS
    const v16bf p_lo = lds_load_a_frag(&ps[wave][0], 0, 64, 0);
    const v16bf p_hi = lds_load_a_frag(&ps[wave][0], 0, 64, 32);

    // ---- oacc += P(16x64) @ vc_chunk(64x64) ----
#pragma unroll
    for (int jd = 0; jd < 4; ++jd) {
      oacc[jd] = wmma_bf16(p_lo, lds_load_b_frag(vts, jd * 16, 64, 0), oacc[jd]);
      oacc[jd] = wmma_bf16(p_hi, lds_load_b_frag(vts, jd * 16, 64, 32), oacc[jd]);
    }
    __syncthreads();
  }

  // ---- normalize, store merged-head layout [B*S][QKV] (bf16) ----
#pragma unroll
  for (int v = 0; v < 8; ++v) {
    const float inv = 1.0f / ssum[v];
    const size_t r = (size_t)qb + wave * 16 + v + rsel;
#pragma unroll
    for (int jd = 0; jd < 4; ++jd)
      attn_out[((size_t)b * S_ + r) * QKV_ + h * 64 + jd * 16 + col] =
          (bf16)(oacc[jd][v] * inv);
  }
}

// ---------------------------------------------------------------------------
extern "C" void kernel_launch(void* const* d_in, const int* in_sizes, int n_in,
                              void* d_out, int out_size, void* d_ws,
                              size_t ws_size, hipStream_t stream) {
  const float* q      = (const float*)d_in[0];
  const float* Wq     = (const float*)d_in[1];
  const float* Wk     = (const float*)d_in[2];
  const float* Wv     = (const float*)d_in[3];
  const float* Wlin   = (const float*)d_in[4];
  const float* conv_w = (const float*)d_in[5];
  const float* logt   = (const float*)d_in[6];

  float* out    = (float*)d_out;                          // [B,S,HID]
  float* logits = (float*)d_out + (size_t)B_ * S_ * HID_; // [B,H,S,SK]

  // workspace layout (bytes); total 72 MB
  char* ws = (char*)d_ws;
  bf16* Qbf    = (bf16*)(ws);                          // 16 MB
  bf16* Kbf    = (bf16*)(ws + (((size_t)16) << 20));   // 16 MB
  bf16* Vbf    = (bf16*)(ws + (((size_t)32) << 20));   // 16 MB
  bf16* kc     = (bf16*)(ws + (((size_t)48) << 20));   //  4 MB
  bf16* vc     = (bf16*)(ws + (((size_t)52) << 20));   //  4 MB
  bf16* attnbf = (bf16*)(ws + (((size_t)56) << 20));   // 16 MB

  const int M = B_ * S_;  // 8192

  dim3 gblk(256);
  dim3 ggrid(M / 128, QKV_ / 128);  // (64, 8)

  // 1) projections (Q scaled by exp(-log_temp) in epilogue)
  gemm_bf16_kernel<true, false><<<ggrid, gblk, 0, stream>>>(
      q, Wq, Qbf, nullptr, M, QKV_, HID_, logt);
  gemm_bf16_kernel<true, false><<<ggrid, gblk, 0, stream>>>(
      q, Wk, Kbf, nullptr, M, QKV_, HID_, nullptr);
  gemm_bf16_kernel<true, false><<<ggrid, gblk, 0, stream>>>(
      q, Wv, Vbf, nullptr, M, QKV_, HID_, nullptr);

  // 2) depthwise conv (both K and V in one pass)
  const int convN = B_ * H_ * SK_ * HD_;  // 2,097,152
  dwconv_kernel<<<convN / 256, 256, 0, stream>>>(Kbf, Vbf, conv_w, kc, vc);

  // 3) fused logits + online softmax + att@vc
  dim3 agrid(S_ / 128, B_ * H_);  // (32, 32)
  attn_kernel<<<agrid, 256, 0, stream>>>(Qbf, kc, vc, logits, attnbf);

  // 4) final linear -> fp32 d_out
  dim3 fgrid(M / 128, HID_ / 128);
  gemm_bf16_kernel<false, true><<<fgrid, gblk, 0, stream>>>(
      attnbf, Wlin, nullptr, out, M, HID_, QKV_, nullptr);
}
